// ComposerModel_5617817223988
// MI455X (gfx1250) — matrix operand order repaired and checked by reading, not compiled, not run on hardware
//
#include <hip/hip_runtime.h>
#include <hip/hip_bf16.h>
#include <math.h>

// Problem constants (match reference)
#define B_   64
#define T_   32
#define V_   32000
#define D_   1024
#define P_   8
#define DEP_ 4
#define M_   (B_ * T_)   // 2048

typedef __bf16 bf16;
typedef __attribute__((ext_vector_type(16))) __bf16 v16bf;
typedef __attribute__((ext_vector_type(8)))  __bf16 bf16x8;
typedef __attribute__((ext_vector_type(4)))  __bf16 bf16x4;
typedef __attribute__((ext_vector_type(8)))  float  v8f;
typedef __attribute__((ext_vector_type(4)))  float  f32x4;

// ---------------------------------------------------------------------------
// WMMA fragment helpers (wave32, 16x16x32 bf16, layouts per CDNA5 ISA 7.12.2)
// ---------------------------------------------------------------------------
__device__ __forceinline__ v8f wmma_bf16(v16bf a, v16bf b, v8f c) {
  return __builtin_amdgcn_wmma_f32_16x16x32_bf16(
      /*neg_a=*/false, a, /*neg_b=*/false, b,
      /*c_mod=*/(short)0, c, /*reuse_a=*/false, /*reuse_b=*/false);
}

// A: 16x32 bf16, row-major [m][k].  lane(m=L%16, g=L/16) holds
// K = {8g..8g+7} in halves 0..7 and K = {16+8g..16+8g+7} in halves 8..15.
__device__ __forceinline__ v16bf load_a_frag(const bf16* A, int lda, int row0, int k0) {
  const int lane = threadIdx.x & 31;
  const int m = lane & 15, g = lane >> 4;
  const bf16* p = A + (size_t)(row0 + m) * lda + (k0 + g * 8);
  bf16x8 lo = *(const bf16x8*)(p);
  bf16x8 hi = *(const bf16x8*)(p + 16);
  v16bf r;
#pragma unroll
  for (int i = 0; i < 8; ++i) { r[i] = lo[i]; r[i + 8] = hi[i]; }
  return r;
}

// B: 32x16 bf16 from pre-transposed BT (row-major [n][k]).
// lane(n=L%16, g=L/16) holds K = {8g..8g+7} in halves 0..7 and K = {16+8g..16+8g+7} in halves 8..15.
__device__ __forceinline__ v16bf load_b_frag(const bf16* BT, int ldb, int col0, int k0) {
  const int lane = threadIdx.x & 31;
  const int n = lane & 15, g = lane >> 4;
  const bf16* p = BT + (size_t)(col0 + n) * ldb + (k0 + g * 8);
  bf16x8 lo = *(const bf16x8*)(p);
  bf16x8 hi = *(const bf16x8*)(p + 16);
  v16bf r;
#pragma unroll
  for (int i = 0; i < 8; ++i) { r[i] = lo[i]; r[i + 8] = hi[i]; }
  return r;
}

// 2x2 register-tiled WMMA core: each wave owns a 32x32 output patch.
// acc layout: acc[ti*2+tj] is the 16x16 tile at (row0+16*ti, col0+16*tj).
__device__ __forceinline__ void gemm_core_2x2(const bf16* __restrict__ A, int lda,
                                              const bf16* __restrict__ BT, int ldb,
                                              int row0, int col0, v8f acc[4]) {
#pragma unroll 2
  for (int k = 0; k < D_; k += 32) {
    v16bf a0 = load_a_frag(A, lda, row0,      k);
    v16bf a1 = load_a_frag(A, lda, row0 + 16, k);
    v16bf b0 = load_b_frag(BT, ldb, col0,      k);
    v16bf b1 = load_b_frag(BT, ldb, col0 + 16, k);
    acc[0] = wmma_bf16(a0, b0, acc[0]);
    acc[1] = wmma_bf16(a0, b1, acc[1]);
    acc[2] = wmma_bf16(a1, b0, acc[2]);
    acc[3] = wmma_bf16(a1, b1, acc[3]);
  }
}

// ---------------------------------------------------------------------------
// 1) Embedding: h = emb[x] + pos   (f32 + bf16 mirrors); zero div accumulator
// ---------------------------------------------------------------------------
__global__ __launch_bounds__(256) void k_embed(const int* __restrict__ x,
                                               const float* __restrict__ emb,
                                               const float* __restrict__ pos,
                                               float* __restrict__ h,
                                               bf16* __restrict__ h_bf,
                                               float* __restrict__ div_out) {
  const int m = blockIdx.x;                 // 0..2047
  const int t = m & (T_ - 1);
  const int tok = x[m];
  const int base = threadIdx.x * 4;         // 256*4 == 1024 == D
  const float* e  = emb + (size_t)tok * D_;
  const float* pp = pos + (size_t)t * D_;
#pragma unroll
  for (int i = 0; i < 4; ++i) {
    const int d = base + i;
    const float v = e[d] + pp[d];
    h[(size_t)m * D_ + d]    = v;
    h_bf[(size_t)m * D_ + d] = (bf16)v;
  }
  if (m == 0 && threadIdx.x == 0) *div_out = 0.0f;
}

// ---------------------------------------------------------------------------
// 2a) Transpose + convert w1,w2 -> bf16 BT layout: bt[p][e][d] = w[p][d][e]
// ---------------------------------------------------------------------------
__global__ __launch_bounds__(256) void k_transpose_bf(const float* __restrict__ w1,
                                                      const float* __restrict__ w2,
                                                      bf16* __restrict__ bt1,
                                                      bf16* __restrict__ bt2) {
  __shared__ float tile[32][33];
  const int pz = blockIdx.z;                      // 0..15
  const float* src = (pz < P_) ? w1 : w2;
  bf16* dst = (pz < P_) ? bt1 : bt2;
  const int p = pz & (P_ - 1);
  const int d0 = blockIdx.x * 32;
  const int e0 = blockIdx.y * 32;
  const int tx = threadIdx.x & 31, ty = threadIdx.x >> 5;   // 32 x 8
  const float* s = src + (size_t)p * D_ * D_;
#pragma unroll
  for (int i = 0; i < 4; ++i) {
    const int r = ty + i * 8;
    tile[r][tx] = s[(size_t)(d0 + r) * D_ + (e0 + tx)];
  }
  __syncthreads();
  bf16* o = dst + (size_t)p * D_ * D_;
#pragma unroll
  for (int i = 0; i < 4; ++i) {
    const int r = ty + i * 8;
    o[(size_t)(e0 + r) * D_ + (d0 + tx)] = (bf16)tile[tx][r];
  }
}

// 2b) head_w [V][D] is already the BT layout; just convert f32 -> bf16 once.
__global__ __launch_bounds__(256) void k_conv_head(const float* __restrict__ hw,
                                                   bf16* __restrict__ out) {
  const size_t i = ((size_t)blockIdx.x * 256 + threadIdx.x) * 4;
  f32x4 v = *(const f32x4*)(hw + i);
  bf16x4 o;
#pragma unroll
  for (int j = 0; j < 4; ++j) o[j] = (bf16)v[j];
  *(bf16x4*)(out + i) = o;
}

// ---------------------------------------------------------------------------
// 3) Selector: ctx = mean_t h, logits = ctx @ sel_w[d]^T + sel_b, gumbel softmax
// ---------------------------------------------------------------------------
__global__ __launch_bounds__(256) void k_sel(const float* __restrict__ h,
                                             const float* __restrict__ sel_w,
                                             const float* __restrict__ sel_b,
                                             const float* __restrict__ gumbel,
                                             float* __restrict__ weights, int dep) {
  __shared__ float red[256][P_];
  const int b = blockIdx.x;
  const int tid = threadIdx.x;
  const int base = tid * 4;
  float ctx[4] = {0.f, 0.f, 0.f, 0.f};
  for (int t = 0; t < T_; ++t) {
    const float* row = h + (size_t)(b * T_ + t) * D_ + base;
#pragma unroll
    for (int i = 0; i < 4; ++i) ctx[i] += row[i];
  }
  const float invT = 1.0f / (float)T_;
#pragma unroll
  for (int i = 0; i < 4; ++i) ctx[i] *= invT;
#pragma unroll
  for (int p = 0; p < P_; ++p) {
    const float* sw = sel_w + ((size_t)dep * P_ + p) * D_ + base;
    float s = 0.f;
#pragma unroll
    for (int i = 0; i < 4; ++i) s += ctx[i] * sw[i];
    red[tid][p] = s;
  }
  __syncthreads();
  for (int off = 128; off > 0; off >>= 1) {
    if (tid < off) {
#pragma unroll
      for (int p = 0; p < P_; ++p) red[tid][p] += red[tid + off][p];
    }
    __syncthreads();
  }
  if (tid == 0) {
    float lg[P_];
    float mx = -1e30f;
#pragma unroll
    for (int p = 0; p < P_; ++p) {
      lg[p] = red[0][p] + sel_b[dep * P_ + p] +
              gumbel[((size_t)dep * B_ + b) * P_ + p];   // TAU == 1
      mx = fmaxf(mx, lg[p]);
    }
    float sum = 0.f;
#pragma unroll
    for (int p = 0; p < P_; ++p) { lg[p] = __expf(lg[p] - mx); sum += lg[p]; }
    const float inv = 1.0f / sum;
#pragma unroll
    for (int p = 0; p < P_; ++p) weights[b * P_ + p] = lg[p] * inv;
  }
}

// KL(uniform || avg) accumulation into div slot (sequential launches -> deterministic)
__global__ void k_div(const float* __restrict__ weights, float* __restrict__ div_out) {
  __shared__ float avg[P_];
  const int tid = threadIdx.x;   // 64 threads
  if (tid < P_) {
    float s = 0.f;
    for (int b = 0; b < B_; ++b) s += weights[b * P_ + tid];
    avg[tid] = s * (1.0f / (float)B_);
  }
  __syncthreads();
  if (tid == 0) {
    const float u = 1.0f / (float)P_;
    float acc = 0.f;
#pragma unroll
    for (int p = 0; p < P_; ++p) acc += u * (logf(u) - logf(avg[p]));
    *div_out += acc;
  }
}

// ---------------------------------------------------------------------------
// 4) GEMM1: u1[p] = gelu(h @ w1[p] + b1[p])   (WMMA bf16, f32 accum)
//    block = 8 waves (2x4), 64x128 tile; wave = 32x32; grid (M/64, D/128, P)
// ---------------------------------------------------------------------------
__global__ __launch_bounds__(256) void k_gemm1(const bf16* __restrict__ h_bf,
                                               const bf16* __restrict__ bt1,
                                               const float* __restrict__ b1,
                                               bf16* __restrict__ u1) {
  const int w = threadIdx.x >> 5;
  const int row0 = blockIdx.x * 64 + (w & 1) * 32;
  const int col0 = blockIdx.y * 128 + (w >> 1) * 32;
  const int p = blockIdx.z;
  const bf16* BT = bt1 + (size_t)p * D_ * D_;
  v8f acc[4] = {v8f{}, v8f{}, v8f{}, v8f{}};
  gemm_core_2x2(h_bf, D_, BT, D_, row0, col0, acc);

  const int lane = threadIdx.x & 31;
  const int n = lane & 15, g = lane >> 4;
  bf16* out = u1 + (size_t)p * M_ * D_;
#pragma unroll
  for (int ti = 0; ti < 2; ++ti) {
#pragma unroll
    for (int tj = 0; tj < 2; ++tj) {
      const int col = col0 + tj * 16 + n;
      const float bias = b1[p * D_ + col];
      const v8f c = acc[ti * 2 + tj];
#pragma unroll
      for (int j = 0; j < 8; ++j) {
        const float xv = c[j] + bias;
        const float ge = 0.5f * xv * (1.0f + erff(xv * 0.70710678118f)); // exact gelu
        out[(size_t)(row0 + ti * 16 + g * 8 + j) * D_ + col] = (bf16)ge;
      }
    }
  }
}

// ---------------------------------------------------------------------------
// 5) GEMM2 + weighted residual merge:
//    h += sum_p weights[b,p] * (u1[p] @ w2[p] + b2[p]);  wave tile = 32 rows
//    == exactly one batch entry (T==32) -> scalar weight per wave tile.
// ---------------------------------------------------------------------------
__global__ __launch_bounds__(256) void k_gemm2(const bf16* __restrict__ u1,
                                               const bf16* __restrict__ bt2,
                                               const float* __restrict__ b2,
                                               const float* __restrict__ weights,
                                               float* __restrict__ h,
                                               bf16* __restrict__ h_bf) {
  const int w = threadIdx.x >> 5;
  const int row0 = blockIdx.x * 64 + (w & 1) * 32;
  const int col0 = blockIdx.y * 128 + (w >> 1) * 32;
  const int b = row0 / T_;                 // row0 is a multiple of 32 == T
  const int lane = threadIdx.x & 31;
  const int n = lane & 15, g = lane >> 4;

  float facc[4][8];
#pragma unroll
  for (int q = 0; q < 4; ++q)
#pragma unroll
    for (int j = 0; j < 8; ++j) facc[q][j] = 0.f;

  for (int p = 0; p < P_; ++p) {
    const bf16* A  = u1 + (size_t)p * M_ * D_;
    const bf16* BT = bt2 + (size_t)p * D_ * D_;
    v8f acc[4] = {v8f{}, v8f{}, v8f{}, v8f{}};
    gemm_core_2x2(A, D_, BT, D_, row0, col0, acc);
    const float wgt = weights[b * P_ + p];
#pragma unroll
    for (int ti = 0; ti < 2; ++ti) {
#pragma unroll
      for (int tj = 0; tj < 2; ++tj) {
        const float bias = b2[p * D_ + col0 + tj * 16 + n];
        const v8f c = acc[ti * 2 + tj];
#pragma unroll
        for (int j = 0; j < 8; ++j) facc[ti * 2 + tj][j] += wgt * (c[j] + bias);
      }
    }
  }
#pragma unroll
  for (int ti = 0; ti < 2; ++ti) {
#pragma unroll
    for (int tj = 0; tj < 2; ++tj) {
#pragma unroll
      for (int j = 0; j < 8; ++j) {
        const size_t idx =
            (size_t)(row0 + ti * 16 + g * 8 + j) * D_ + (col0 + tj * 16 + n);
        const float v = h[idx] + facc[ti * 2 + tj][j];
        h[idx]    = v;
        h_bf[idx] = (bf16)v;
      }
    }
  }
}

// ---------------------------------------------------------------------------
// 6) LayerNorm -> bf16
// ---------------------------------------------------------------------------
__global__ __launch_bounds__(256) void k_ln(const float* __restrict__ h,
                                            const float* __restrict__ gam,
                                            const float* __restrict__ bet,
                                            bf16* __restrict__ hn_bf) {
  __shared__ float s1[256], s2[256];
  const int m = blockIdx.x, tid = threadIdx.x;
  const int base = tid * 4;
  const float* row = h + (size_t)m * D_;
  float v[4];
  float s = 0.f, q = 0.f;
#pragma unroll
  for (int i = 0; i < 4; ++i) { v[i] = row[base + i]; s += v[i]; q += v[i] * v[i]; }
  s1[tid] = s; s2[tid] = q;
  __syncthreads();
  for (int off = 128; off > 0; off >>= 1) {
    if (tid < off) { s1[tid] += s1[tid + off]; s2[tid] += s2[tid + off]; }
    __syncthreads();
  }
  const float mu  = s1[0] * (1.0f / (float)D_);
  const float var = s2[0] * (1.0f / (float)D_) - mu * mu;
  const float rs  = rsqrtf(var + 1e-5f);
#pragma unroll
  for (int i = 0; i < 4; ++i) {
    const int d = base + i;
    hn_bf[(size_t)m * D_ + d] = (bf16)((v[i] - mu) * rs * gam[d] + bet[d]);
  }
}

// ---------------------------------------------------------------------------
// 7) Vocab head: logits = hn @ head_w^T  (head_bf pre-converted, BT layout,
//    L2-resident). block = 64x128, wave = 32x32; grid (M/64, V/128).
// ---------------------------------------------------------------------------
__global__ __launch_bounds__(256) void k_head(const bf16* __restrict__ hn_bf,
                                              const bf16* __restrict__ head_bf,
                                              float* __restrict__ out) {
  const int w = threadIdx.x >> 5;
  const int row0 = blockIdx.x * 64 + (w & 1) * 32;
  const int col0 = blockIdx.y * 128 + (w >> 1) * 32;
  const int lane = threadIdx.x & 31;
  const int n = lane & 15, g = lane >> 4;

  v8f acc[4] = {v8f{}, v8f{}, v8f{}, v8f{}};
#pragma unroll 2
  for (int k = 0; k < D_; k += 32) {
    // hint: pull the next B chunk toward the caches (global_prefetch_b8)
    if (k + 32 < D_) {
      __builtin_prefetch(head_bf + (size_t)(col0 + n) * D_ + k + 32 + g * 16, 0, 1);
    }
    v16bf a0 = load_a_frag(hn_bf, D_, row0,      k);
    v16bf a1 = load_a_frag(hn_bf, D_, row0 + 16, k);
    v16bf b0 = load_b_frag(head_bf, D_, col0,      k);
    v16bf b1 = load_b_frag(head_bf, D_, col0 + 16, k);
    acc[0] = wmma_bf16(a0, b0, acc[0]);
    acc[1] = wmma_bf16(a0, b1, acc[1]);
    acc[2] = wmma_bf16(a1, b0, acc[2]);
    acc[3] = wmma_bf16(a1, b1, acc[3]);
  }
#pragma unroll
  for (int ti = 0; ti < 2; ++ti) {
#pragma unroll
    for (int tj = 0; tj < 2; ++tj) {
      const v8f c = acc[ti * 2 + tj];
#pragma unroll
      for (int j = 0; j < 8; ++j) {
        out[(size_t)(row0 + ti * 16 + g * 8 + j) * V_ + (col0 + tj * 16 + n)] = c[j];
      }
    }
  }
}

// ---------------------------------------------------------------------------
// Launch
// ---------------------------------------------------------------------------
extern "C" void kernel_launch(void* const* d_in, const int* in_sizes, int n_in,
                              void* d_out, int out_size, void* d_ws, size_t ws_size,
                              hipStream_t stream) {
  (void)in_sizes; (void)n_in; (void)out_size; (void)ws_size;
  const int*   x      = (const int*)d_in[0];
  const float* emb    = (const float*)d_in[1];
  const float* pos    = (const float*)d_in[2];
  const float* w1     = (const float*)d_in[3];
  const float* b1     = (const float*)d_in[4];
  const float* w2     = (const float*)d_in[5];
  const float* b2     = (const float*)d_in[6];
  const float* sel_w  = (const float*)d_in[7];
  const float* sel_b  = (const float*)d_in[8];
  const float* ln_g   = (const float*)d_in[9];
  const float* ln_b   = (const float*)d_in[10];
  const float* head_w = (const float*)d_in[11];
  const float* gumbel = (const float*)d_in[12];

  float* out_logits = (float*)d_out;                       // [B,T,V]
  float* out_div    = out_logits + (size_t)M_ * V_;        // scalar tail

  // Workspace carve-up (all chunks 256B-aligned; total ~149 MB)
  char* ws = (char*)d_ws;
  size_t off = 0;
  float* h       = (float*)(ws + off); off += (size_t)M_ * D_ * 4;        // 8 MB
  bf16*  h_bf    = (bf16*) (ws + off); off += (size_t)M_ * D_ * 2;        // 4 MB
  bf16*  hn_bf   = (bf16*) (ws + off); off += (size_t)M_ * D_ * 2;        // 4 MB
  bf16*  u1      = (bf16*) (ws + off); off += (size_t)P_ * M_ * D_ * 2;   // 32 MB
  bf16*  bt1     = (bf16*) (ws + off); off += (size_t)P_ * D_ * D_ * 2;   // 16 MB
  bf16*  bt2     = (bf16*) (ws + off); off += (size_t)P_ * D_ * D_ * 2;   // 16 MB
  bf16*  head_bf = (bf16*) (ws + off); off += (size_t)V_ * D_ * 2;        // 64 MB
  float* wts     = (float*)(ws + off); off += (size_t)B_ * P_ * 4;        // 2 KB

  k_embed<<<M_, 256, 0, stream>>>(x, emb, pos, h, h_bf, out_div);
  k_transpose_bf<<<dim3(D_ / 32, D_ / 32, 2 * P_), 256, 0, stream>>>(w1, w2, bt1, bt2);
  k_conv_head<<<(int)(((size_t)V_ * D_) / 1024), 256, 0, stream>>>(head_w, head_bf);

  for (int dep = 0; dep < DEP_; ++dep) {
    k_sel<<<B_, 256, 0, stream>>>(h, sel_w, sel_b, gumbel, wts, dep);
    k_div<<<1, 64, 0, stream>>>(wts, out_div);
    k_gemm1<<<dim3(M_ / 64, D_ / 128, P_), 256, 0, stream>>>(h_bf, bt1, b1, u1);
    k_gemm2<<<dim3(M_ / 64, D_ / 128), 256, 0, stream>>>(u1, bt2, b2, wts, h, h_bf);
  }

  k_ln<<<M_, 256, 0, stream>>>(h, ln_g, ln_b, hn_bf);
  k_head<<<dim3(M_ / 64, V_ / 128), 256, 0, stream>>>(hn_bf, head_bf, out_logits);
}
